// MultiHeadAttention_60378650247490
// MI455X (gfx1250) — matrix operand order repaired and checked by reading, not compiled, not run on hardware
//
#include <hip/hip_runtime.h>
#include <hip/hip_bf16.h>
#include <cstdint>

// ---- problem constants (reference: B=2, L=1024, D=1024, H=16, dk=64) ----
#define Bn 2
#define Ln 1024
#define Dn 1024
#define Hn 16
#define DKn 64

typedef __attribute__((ext_vector_type(16))) _Float16 v16h;
typedef __attribute__((ext_vector_type(8)))  _Float16 v8h;
typedef __attribute__((ext_vector_type(4)))  _Float16 v4h;
typedef __attribute__((ext_vector_type(8)))  float    v8f;
typedef __attribute__((ext_vector_type(4)))  unsigned int u32x4;
typedef __attribute__((ext_vector_type(8)))  int      i32x8;
typedef __attribute__((ext_vector_type(4)))  int      i32x4;

// ---------------------------------------------------------------------------
// WMMA helpers (CDNA5 wave32 layouts, 16x16x32 f16 -> f32)
// ---------------------------------------------------------------------------
__device__ __forceinline__ v8f wmma32(v16h a, v16h b, v8f c) {
  return __builtin_amdgcn_wmma_f32_16x16x32_f16(
      /*neg_a=*/false, a, /*neg_b=*/false, b,
      /*c_mod=*/(short)0, c, /*reuse_a=*/false, /*reuse_b=*/false);
}

// A fragment (16x32 f16, row-major source; A pre-offset to tile row 0, col 0).
// Lane = row M=lane%16; VGPR0..7 = K kb..kb+15 (contiguous),
// kb = (lane/16)*16. Works for global or LDS pointers.
__device__ __forceinline__ v16h frag_a(const _Float16* A, int lda, int k0) {
  int lane = threadIdx.x & 31;
  const _Float16* p = A + (size_t)(lane & 15) * lda + k0 + ((lane >> 4) << 4);
  v8h lo = *(const v8h*)p;
  v8h hi = *(const v8h*)(p + 8);
  v16h a;
#pragma unroll
  for (int i = 0; i < 8; ++i) { a[i] = lo[i]; a[i + 8] = hi[i]; }
  return a;
}

// B fragment where B[k,n] = T[n,k] (T row-major f16; pre-offset to row n0).
// Lane = column N=lane%16; K range = k0 + (lane/16)*16 .. +15 (contiguous).
__device__ __forceinline__ v16h frag_bT_f16(const _Float16* T, int ldt, int k0) {
  int lane = threadIdx.x & 31;
  const _Float16* p = T + (size_t)(lane & 15) * ldt + k0 + ((lane >> 4) << 4);
  v8h lo = *(const v8h*)p;
  v8h hi = *(const v8h*)(p + 8);
  v16h b;
#pragma unroll
  for (int i = 0; i < 8; ++i) { b[i] = lo[i]; b[i + 8] = hi[i]; }
  return b;
}

// B fragment from a row-major K x N source S (e.g. V tile): B[k,n] = S[k,n].
__device__ __forceinline__ v16h frag_b_rowmajor(const _Float16* S, int lds_, int k0) {
  int lane = threadIdx.x & 31;
  const _Float16* p = S + (size_t)(k0 + ((lane >> 4) << 4)) * lds_ + (lane & 15);
  v16h b;
#pragma unroll
  for (int j = 0; j < 16; ++j) b[j] = p[(size_t)j * lds_];
  return b;
}

// A fragment from f32 LDS (softmax probabilities), converted to f16.
__device__ __forceinline__ v16h frag_a_lds_f32(const float* S, int lds_, int k0) {
  int lane = threadIdx.x & 31;
  const float* p = S + (lane & 15) * lds_ + k0 + ((lane >> 4) << 4);
  v16h a;
#pragma unroll
  for (int i = 0; i < 8; ++i) {
    a[i]     = (_Float16)p[i];
    a[i + 8] = (_Float16)p[i + 8];
  }
  return a;
}

// ---------------------------------------------------------------------------
// Stage a 16 x K f16 A-tile (row-major, global) into LDS.
// Preferred path: Tensor Data Mover (tensor_load_to_lds), 2D descriptor:
//   tile = K x 16 elements of 2 bytes, tensor = K x Mtot, stride K.
// Fallback: cooperative vector loads.
// Must be called by all 256 threads; ends fully synchronized.
// ---------------------------------------------------------------------------
__device__ __forceinline__ void stage_a_tile(_Float16* sA,
                                             const _Float16* Aglobal,
                                             int K, int Mtot) {
#if defined(__has_builtin) && __has_builtin(__builtin_amdgcn_tensor_load_to_lds)
  const int wave = threadIdx.x >> 5;
  if (wave == 0) {
    const uint32_t lds_off = (uint32_t)(uintptr_t)sA;          // LDS byte addr
    const uint64_t gaddr   = (uint64_t)(uintptr_t)Aglobal;     // tile start
    u32x4 g0;
    g0[0] = 1u;                                                // count=1
    g0[1] = lds_off;                                           // lds_addr
    g0[2] = (uint32_t)gaddr;                                   // global_addr lo
    g0[3] = (uint32_t)((gaddr >> 32) & 0x1FFFFFFu) | (2u << 30); // hi | type=2
    i32x8 g1;
    g1[0] = (int)(1u << 16);            // wg_mask=0, data_size=1 (2 bytes)
    g1[1] = (int)((uint32_t)K << 16);   // tensor_dim0[15:0] in bits[31:16]
    g1[2] = (int)((uint32_t)Mtot << 16);// tensor_dim0 hi=0 | tensor_dim1 lo<<16
    g1[3] = (int)((uint32_t)K << 16);   // tensor_dim1 hi=0 | tile_dim0=K
    g1[4] = 16;                         // tile_dim1=16 rows, tile_dim2=0
    g1[5] = K;                          // tensor_dim0_stride[31:0]
    g1[6] = 0;                          // stride hi | dim1_stride lo
    g1[7] = 0;
    i32x4 g2 = {0, 0, 0, 0};
    i32x4 g3 = {0, 0, 0, 0};
#if __clang_major__ >= 23
    i32x8 g4 = {0, 0, 0, 0, 0, 0, 0, 0};
    __builtin_amdgcn_tensor_load_to_lds(g0, g1, g2, g3, g4, 0);
#else
    __builtin_amdgcn_tensor_load_to_lds(g0, g1, g2, g3, 0);
#endif
#if __has_builtin(__builtin_amdgcn_s_wait_tensorcnt)
    __builtin_amdgcn_s_wait_tensorcnt(0);
#endif
  }
  __syncthreads();
#else
  const int t = threadIdx.x;
  const int chunks = 16 * (K / 8);      // v8h chunks in the tile
  for (int i = t; i < chunks; i += 256)
    ((v8h*)sA)[i] = ((const v8h*)Aglobal)[i];
  __syncthreads();
#endif
}

// ---------------------------------------------------------------------------
// Kernel 0: bulk f32 -> f16 conversion (weight pre-pack)
// ---------------------------------------------------------------------------
__global__ void f32_to_f16_kernel(const float* __restrict__ src,
                                  _Float16* __restrict__ dst, int n) {
  const int i = (blockIdx.x * 256 + threadIdx.x) * 8;
  if (i >= n) return;
  float4 a = ((const float4*)(src + i))[0];
  float4 b = ((const float4*)(src + i))[1];
  v8h o;
  o[0] = (_Float16)a.x; o[1] = (_Float16)a.y;
  o[2] = (_Float16)a.z; o[3] = (_Float16)a.w;
  o[4] = (_Float16)b.x; o[5] = (_Float16)b.y;
  o[6] = (_Float16)b.z; o[7] = (_Float16)b.w;
  *(v8h*)(dst + i) = o;
}

// ---------------------------------------------------------------------------
// Kernel 1: pre-LN of query -> qn_h (f16), plus raw query -> x_h (f16 context)
// ---------------------------------------------------------------------------
__global__ void ln_prep_kernel(const float* __restrict__ query,
                               const float* __restrict__ gamma,
                               const float* __restrict__ beta,
                               _Float16* __restrict__ qn_h,
                               _Float16* __restrict__ x_h) {
  const int row = blockIdx.x;            // 0 .. B*L-1
  const int t = threadIdx.x;             // 0 .. 255 ; 4 elems each
  const float* x = query + (size_t)row * Dn;
  float4 v = ((const float4*)x)[t];
  float s  = v.x + v.y + v.z + v.w;
  float sq = v.x * v.x + v.y * v.y + v.z * v.z + v.w * v.w;

  __shared__ float rs[256];
  __shared__ float rq[256];
  rs[t] = s; rq[t] = sq;
  __syncthreads();
#pragma unroll
  for (int off = 128; off > 0; off >>= 1) {
    if (t < off) { rs[t] += rs[t + off]; rq[t] += rq[t + off]; }
    __syncthreads();
  }
  const float mean = rs[0] * (1.0f / Dn);
  const float var  = rq[0] * (1.0f / Dn) - mean * mean;
  const float rstd = rsqrtf(var + 1e-6f);

  float4 g  = ((const float4*)gamma)[t];
  float4 be = ((const float4*)beta)[t];
  v4h qn, xh;
  qn[0] = (_Float16)((v.x - mean) * rstd * g.x + be.x);
  qn[1] = (_Float16)((v.y - mean) * rstd * g.y + be.y);
  qn[2] = (_Float16)((v.z - mean) * rstd * g.z + be.z);
  qn[3] = (_Float16)((v.w - mean) * rstd * g.w + be.w);
  xh[0] = (_Float16)v.x; xh[1] = (_Float16)v.y;
  xh[2] = (_Float16)v.z; xh[3] = (_Float16)v.w;
  *(v4h*)(qn_h + (size_t)row * Dn + t * 4) = qn;
  *(v4h*)(x_h  + (size_t)row * Dn + t * 4) = xh;
}

// ---------------------------------------------------------------------------
// Kernel 2: projection GEMM  C_h[m,n] = f16( sum_k A_h[m,k]*Wh[n,k] + bias[n] )
// A tile (16 x K) staged in LDS by the TDM; all 8 waves of the block share it.
// Each wave owns one 16x16 output tile (8 n-tiles per block).
// ---------------------------------------------------------------------------
__global__ void gemm_proj_kernel(const _Float16* __restrict__ A,
                                 const _Float16* __restrict__ Wh,
                                 const float* __restrict__ bias,
                                 _Float16* __restrict__ C,
                                 int M, int N, int K) {
  __shared__ _Float16 sA[16 * Dn];       // 32 KB
  const int wave = threadIdx.x >> 5;
  const int lane = threadIdx.x & 31;
  const int m0 = (blockIdx.x >> 3) << 4;
  const int n0 = (((blockIdx.x & 7) << 3) + wave) << 4;

  stage_a_tile(sA, A + (size_t)m0 * K, K, M);

  const _Float16* Wbase = Wh + (size_t)n0 * K;
  v8f c = {};
  for (int k0 = 0; k0 < K; k0 += 32) {
    v16h a = frag_a(sA, K, k0);          // ds_load_b128 from LDS
    v16h b = frag_bT_f16(Wbase, K, k0);
    c = wmma32(a, b, c);
  }
  const int col = n0 + (lane & 15);
  const float bv = bias[col];
  const int rbase = m0 + ((lane >> 4) << 3);
#pragma unroll
  for (int r = 0; r < 8; ++r)
    C[(size_t)(rbase + r) * N + col] = (_Float16)(c[r] + bv);
}

// ---------------------------------------------------------------------------
// Kernel 3: fused attention for one (b, h, 16-query-row tile)
//   logits (WMMA QK^T + biases + mask) -> LDS -> softmax (wave shuffles)
//   -> attn written to d_out, probs reused from LDS as WMMA A for attn @ V
// 256 threads = 8 waves. LDS: 16 x 1024 f32 = 64 KB.
// ---------------------------------------------------------------------------
__global__ void attn_kernel(const _Float16* __restrict__ Qh,
                            const _Float16* __restrict__ Kh,
                            const _Float16* __restrict__ Vh,
                            const float* __restrict__ pos_bias,
                            const float* __restrict__ postag_bias,
                            const float* __restrict__ lex_mask,
                            const int* __restrict__ mask,
                            float* __restrict__ attn_out,
                            _Float16* __restrict__ Ah) {
  __shared__ float sP[16 * Ln];          // 64 KB row block of logits/probs

  const int qt  = blockIdx.x % (Ln / 16);
  const int tmp = blockIdx.x / (Ln / 16);
  const int h = tmp % Hn;
  const int b = tmp / Hn;
  const int q0 = qt << 4;
  const int wave = threadIdx.x >> 5;
  const int lane = threadIdx.x & 31;

  // ---- phase 1: logits = Q K^T / sqrt(dk) + pos + postag + lex, masked ----
  const _Float16* Qbase = Qh + (size_t)(b * Ln + q0) * Dn + h * DKn;
  v16h a0 = frag_a(Qbase, Dn, 0);
  v16h a1 = frag_a(Qbase, Dn, 32);

  for (int j = 0; j < 8; ++j) {
    const int n0 = wave * 128 + j * 16;
    const _Float16* Kbase = Kh + (size_t)(b * Ln + n0) * Dn + h * DKn;
    v16h b0 = frag_bT_f16(Kbase, Dn, 0);
    v16h b1 = frag_bT_f16(Kbase, Dn, 32);
    v8f c = {};
    c = wmma32(a0, b0, c);
    c = wmma32(a1, b1, c);

    const int n = n0 + (lane & 15);
    const float lex = lex_mask[b * Ln + n];
    const int   mk  = mask[b * Ln + n];
    const float* pbase = pos_bias    + ((size_t)h * Ln + q0) * Ln + n;
    const float* tbase = postag_bias + (((size_t)(b * Hn + h)) * Ln + q0) * Ln + n;
    __builtin_prefetch(tbase + 8 * Ln, 0, 1);   // global_prefetch_b8
    const int rbase = (lane >> 4) << 3;
#pragma unroll
    for (int r = 0; r < 8; ++r) {
      const int q = rbase + r;
      float v = c[r] * 0.125f /* 1/sqrt(64) */
              + pbase[(size_t)q * Ln] + tbase[(size_t)q * Ln] + lex;
      if (mk == 0) v = -__builtin_inff();
      sP[q * Ln + n] = v;
    }
  }
  __syncthreads();

  // ---- phase 2: softmax; wave w handles rows 2w, 2w+1 (wave32 shuffles) ----
  for (int rr = 0; rr < 2; ++rr) {
    const int q = wave * 2 + rr;
    float* row = sP + q * Ln;
    float vals[32];
    float m = -__builtin_inff();
#pragma unroll
    for (int i = 0; i < 32; ++i) { vals[i] = row[lane + i * 32]; m = fmaxf(m, vals[i]); }
#pragma unroll
    for (int off = 16; off > 0; off >>= 1) m = fmaxf(m, __shfl_xor(m, off, 32));
    float s = 0.0f;
#pragma unroll
    for (int i = 0; i < 32; ++i) { vals[i] = __expf(vals[i] - m); s += vals[i]; }
#pragma unroll
    for (int off = 16; off > 0; off >>= 1) s += __shfl_xor(s, off, 32);
    const float inv = 1.0f / s;
    float* arow = attn_out + (((size_t)(b * Hn + h)) * Ln + q0 + q) * Ln;
#pragma unroll
    for (int i = 0; i < 32; ++i) {
      const float p = vals[i] * inv;
      row[lane + i * 32] = p;
      arow[lane + i * 32] = p;
    }
  }
  __syncthreads();

  // ---- phase 3: head output = probs @ V  (waves 0..3, one 16x16 tile each) --
  if (wave < 4) {
    const int n0 = wave * 16;
    const _Float16* Vbase = Vh + (size_t)b * Ln * Dn + h * DKn + n0;
    v8f c = {};
    for (int k0 = 0; k0 < Ln; k0 += 32) {
      v16h a = frag_a_lds_f32(sP, Ln, k0);
      v16h bb = frag_b_rowmajor(Vbase, Dn, k0);
      c = wmma32(a, bb, c);
    }
    const int col = h * DKn + n0 + (lane & 15);
    const int rbase = b * Ln + q0 + ((lane >> 4) << 3);
#pragma unroll
    for (int r = 0; r < 8; ++r)
      Ah[(size_t)(rbase + r) * Dn + col] = (_Float16)c[r];
  }
}

// ---------------------------------------------------------------------------
// Kernel 4: output projection + bias + residual (f32 out), LDS-staged A tile
// ---------------------------------------------------------------------------
__global__ void gemm_out_kernel(const _Float16* __restrict__ A,
                                const _Float16* __restrict__ Woh,
                                const float* __restrict__ bo,
                                const float* __restrict__ residual,
                                float* __restrict__ out,
                                int M, int N, int K) {
  __shared__ _Float16 sA[16 * Dn];       // 32 KB
  const int wave = threadIdx.x >> 5;
  const int lane = threadIdx.x & 31;
  const int m0 = (blockIdx.x >> 3) << 4;
  const int n0 = (((blockIdx.x & 7) << 3) + wave) << 4;

  stage_a_tile(sA, A + (size_t)m0 * K, K, M);

  const _Float16* Wbase = Woh + (size_t)n0 * K;
  v8f c = {};
  for (int k0 = 0; k0 < K; k0 += 32) {
    v16h a = frag_a(sA, K, k0);
    v16h b = frag_bT_f16(Wbase, K, k0);
    c = wmma32(a, b, c);
  }
  const int col = n0 + (lane & 15);
  const float bv = bo[col];
  const int rbase = m0 + ((lane >> 4) << 3);
#pragma unroll
  for (int r = 0; r < 8; ++r) {
    const size_t idx = (size_t)(rbase + r) * N + col;
    out[idx] = c[r] + bv + residual[idx];
  }
}

// ---------------------------------------------------------------------------
// launch
// ---------------------------------------------------------------------------
extern "C" void kernel_launch(void* const* d_in, const int* in_sizes, int n_in,
                              void* d_out, int out_size, void* d_ws, size_t ws_size,
                              hipStream_t stream) {
  const float* query   = (const float*)d_in[0];
  const float* posb    = (const float*)d_in[1];
  const float* postagb = (const float*)d_in[2];
  const float* lexm    = (const float*)d_in[3];
  const int*   mask    = (const int*)d_in[4];
  const float* Wq = (const float*)d_in[5];
  const float* bq = (const float*)d_in[6];
  const float* Wk = (const float*)d_in[7];
  const float* bk = (const float*)d_in[8];
  const float* Wv = (const float*)d_in[9];
  const float* bv = (const float*)d_in[10];
  const float* Wo = (const float*)d_in[11];
  const float* bo = (const float*)d_in[12];
  const float* gamma = (const float*)d_in[13];
  const float* beta  = (const float*)d_in[14];

  float* out_ptr  = (float*)d_out;                       // (B,L,D)
  float* attn_ptr = out_ptr + (size_t)Bn * Ln * Dn;      // (B,H,L,L)

  // f16 scratch partitions: 6 activations (4 MB each) + 4 weights (2 MB each)
  const size_t SZ = (size_t)Bn * Ln * Dn;                // activation elements
  const size_t WZ = (size_t)Dn * Dn;                     // weight elements
  _Float16* qn_h = (_Float16*)d_ws;
  _Float16* x_h  = qn_h + SZ;
  _Float16* Qh   = x_h + SZ;
  _Float16* Kh   = Qh + SZ;
  _Float16* Vh   = Kh + SZ;
  _Float16* Ah   = Vh + SZ;
  _Float16* Wq_h = Ah + SZ;
  _Float16* Wk_h = Wq_h + WZ;
  _Float16* Wv_h = Wk_h + WZ;
  _Float16* Wo_h = Wv_h + WZ;

  const int M = Bn * Ln, N = Dn, K = Dn;

  // 0) weight pre-pack to f16 (each 1M elements, 8 per thread)
  const int cvt_blocks = (int)(WZ / (256 * 8));
  f32_to_f16_kernel<<<cvt_blocks, 256, 0, stream>>>(Wq, Wq_h, (int)WZ);
  f32_to_f16_kernel<<<cvt_blocks, 256, 0, stream>>>(Wk, Wk_h, (int)WZ);
  f32_to_f16_kernel<<<cvt_blocks, 256, 0, stream>>>(Wv, Wv_h, (int)WZ);
  f32_to_f16_kernel<<<cvt_blocks, 256, 0, stream>>>(Wo, Wo_h, (int)WZ);

  // 1) layernorm + f16 conversion
  ln_prep_kernel<<<Bn * Ln, 256, 0, stream>>>(query, gamma, beta, qn_h, x_h);

  // 2) projections: (M/16) row-blocks x 8 blocks of 8 n-tiles
  const int proj_blocks = (M / 16) * 8;
  gemm_proj_kernel<<<proj_blocks, 256, 0, stream>>>(qn_h, Wq_h, bq, Qh, M, N, K);
  gemm_proj_kernel<<<proj_blocks, 256, 0, stream>>>(x_h,  Wk_h, bk, Kh, M, N, K);
  gemm_proj_kernel<<<proj_blocks, 256, 0, stream>>>(x_h,  Wv_h, bv, Vh, M, N, K);

  // 3) fused attention: one block per (b, h, 16-query-row tile)
  attn_kernel<<<Bn * Hn * (Ln / 16), 256, 0, stream>>>(
      Qh, Kh, Vh, posb, postagb, lexm, mask, attn_ptr, Ah);

  // 4) output projection + residual
  gemm_out_kernel<<<proj_blocks, 256, 0, stream>>>(Ah, Wo_h, bo, query, out_ptr,
                                                   M, N, K);
}
